// SIRBetaNet_60739427500583
// MI455X (gfx1250) — matrix operand order repaired and checked by reading, not compiled, not run on hardware
//
#include <hip/hip_runtime.h>
#include <math.h>

// CDNA5 / gfx1250: wave32. Layer-2 GEMM on v_wmma_f32_16x16x32_f16 (fully dense
// M=16,N=16,K=32); layer-1 (K=3) on packed-f16 VALU which lands directly in the
// WMMA B-operand layout (no transpose/swizzle needed).
typedef __attribute__((ext_vector_type(16))) _Float16 v16h;
typedef __attribute__((ext_vector_type(8)))  float    v8f;
typedef __attribute__((ext_vector_type(2)))  _Float16 v2h;

union V16H { v16h v; v2h p[8]; };

#define HORIZON 28
#define GAMMA   0.1f

// v_cvt_pk_rtz_f16_f32
static __device__ __forceinline__ v2h pk2(float a, float b) {
    return __builtin_bit_cast(v2h, __builtin_amdgcn_cvt_pkrtz(a, b));
}
// ds_swizzle SWAPX16: swap lane halves of the wave32
static __device__ __forceinline__ float swapx16(float x) {
    return __int_as_float(__builtin_amdgcn_ds_swizzle(__float_as_int(x), 0x401F));
}
// guaranteed single-instruction relu (src0 = inline-const 0, src1 = VGPR)
static __device__ __forceinline__ float relu1(float x) {
    float r;
    asm("v_max_num_f32 %0, 0, %1" : "=v"(r) : "v"(x));
    return r;
}
static __device__ __forceinline__ v8f wmma_f16(const v16h& a, const v16h& b, const v8f& c) {
    return __builtin_amdgcn_wmma_f32_16x16x32_f16(false, a, false, b, (short)0, c, false, false);
}

__global__ void __launch_bounds__(256)
SIRBetaNet_60739427500583_kernel(const float* __restrict__ in,   // (B, 8, 3)
                                 const float* __restrict__ W1p,  // (3, 32)
                                 const float* __restrict__ b1v,  // (32,)
                                 const float* __restrict__ W2p,  // (32, 16)
                                 const float* __restrict__ b2v,  // (16,)
                                 const float* __restrict__ W3p,  // (16,)
                                 const float* __restrict__ b3p,  // (1,)
                                 float* __restrict__ out,        // (B, 28)
                                 int ntiles) {
    const int lane = threadIdx.x & 31;
    const int m    = lane & 15;          // column n (batch row within tile)
    const int hf   = (lane >> 4) & 1;    // lane half: features 16*hf .. 16*hf+15
    const int tile = blockIdx.x * (blockDim.x >> 5) + (threadIdx.x >> 5);
    if (tile >= ntiles) return;          // wave-uniform; EXEC all-ones inside

    // ---- Constant weights in registers (loaded once) ----
    // Layer-1 weights, packed f16 pairs per lane: feature j = 16*hf + 2v (+1)
    v2h w1r[3][8], b1r[8];
    #pragma unroll
    for (int v = 0; v < 8; ++v) {
        const int f = 16 * hf + 2 * v;
        #pragma unroll
        for (int c = 0; c < 3; ++c)
            w1r[c][v] = pk2(W1p[c * 32 + f], W1p[c * 32 + f + 1]);
        b1r[v] = pk2(b1v[f], b1v[f + 1]);
    }
    // A2 = W2^T as 16x32 f16 A-matrix: A2[m][k] = W2[k][m]
    V16H a2;
    #pragma unroll
    for (int v = 0; v < 8; ++v) {
        const int k0 = 16 * hf + 2 * v;  // contiguous K map (matches B operand)
        a2.p[v] = pk2(W2p[k0 * 16 + m], W2p[(k0 + 1) * 16 + m]);
    }
    // Layer-2 bias into the C accumulator; layer-3 weights per lane
    v8f c2;
    float w3r[8];
    #pragma unroll
    for (int v = 0; v < 8; ++v) {
        c2[v]  = b2v[v + 8 * hf];
        w3r[v] = W3p[v + 8 * hf];
    }
    const float b3h = 0.5f * b3p[0];     // folded into each half's dot partial

    // ---- Initial state: inputs[row, 7, 0..2]; both lane halves track row n ----
    const int row = tile * 16 + m;
    const float* xr = in + (size_t)row * 24 + 21;
    float s = xr[0], iC = xr[1], r = xr[2];

    const v2h z2 = pk2(0.0f, 0.0f);
    float preds[HORIZON];

    #pragma unroll
    for (int t = 0; t < HORIZON; ++t) {
        // Layer 1 in packed f16: h1[f][n] = relu(s*W1[0][f] + i*W1[1][f] + r*W1[2][f] + b1[f])
        // Lane (n,hf) produces features 16*hf..16*hf+15 -> B operand in this file's contiguous k order (A2 matches).
        const v2h s2 = pk2(s, s), i2 = pk2(iC, iC), r2 = pk2(r, r);
        V16H b2m;
        #pragma unroll
        for (int v = 0; v < 8; ++v) {
            v2h h = __builtin_elementwise_fma(w1r[2][v], r2, b1r[v]);
            h     = __builtin_elementwise_fma(w1r[1][v], i2, h);
            h     = __builtin_elementwise_fma(w1r[0][v], s2, h);
            b2m.p[v] = __builtin_elementwise_max(h, z2);   // v_pk_max_num_f16 relu
        }

        // Layer 2: h2^T = W2^T @ h1^T + b2  (one dense 16x16x32 WMMA)
        const v8f d2 = wmma_f16(a2.v, b2m.v, c2);

        // Layer 3: beta_lin = relu(h2) . W3 + b3 ; cross-half reduce (1 ds_swizzle)
        float p = b3h;
        #pragma unroll
        for (int v = 0; v < 8; ++v) p = fmaf(relu1(d2[v]), w3r[v], p);
        const float bl = p + swapx16(p);
        // softplus(x) = relu(x) + ln2 * log2(1 + 2^(-|x|*log2e)); arg of log2 in (1,2]
        // -> raw v_exp_f32 / v_log_f32 are safe (no denorm/range fixup needed)
        const float e    = __builtin_amdgcn_exp2f(-fabsf(bl) * 1.44269504f);
        const float beta = relu1(bl) + 0.69314718f * __builtin_amdgcn_logf(1.0f + e);

        // SIR update (v_rcp_f32 instead of precise divide)
        const float N      = s + iC + r;
        const float rcpN   = __builtin_amdgcn_rcpf(N);
        const float newInf = beta * s * iC * rcpN;
        const float newRec = GAMMA * iC;
        s  = s - newInf;
        iC = iC + newInf - newRec;
        r  = r + newRec;
        preds[t] = iC;
    }

    // 28 f32 preds per row: 7 x b128 stores, rows are 112-byte aligned
    if (lane < 16) {
        float4* o = (float4*)(out + (size_t)row * HORIZON);
        #pragma unroll
        for (int q = 0; q < 7; ++q)
            o[q] = make_float4(preds[4 * q], preds[4 * q + 1],
                               preds[4 * q + 2], preds[4 * q + 3]);
    }
}

extern "C" void kernel_launch(void* const* d_in, const int* in_sizes, int n_in,
                              void* d_out, int out_size, void* d_ws, size_t ws_size,
                              hipStream_t stream) {
    const float* in  = (const float*)d_in[0];
    const float* W1  = (const float*)d_in[1];
    const float* b1  = (const float*)d_in[2];
    const float* W2  = (const float*)d_in[3];
    const float* b2  = (const float*)d_in[4];
    const float* W3  = (const float*)d_in[5];
    const float* b3  = (const float*)d_in[6];
    float* out = (float*)d_out;

    const int B      = in_sizes[0] / 24;   // (B, 8, 3)
    const int ntiles = B / 16;             // one 16-row tile per wave32
    const int wavesPerBlock = 8;           // 256 threads
    const int blocks = (ntiles + wavesPerBlock - 1) / wavesPerBlock;

    SIRBetaNet_60739427500583_kernel<<<blocks, 256, 0, stream>>>(
        in, W1, b1, W2, b2, W3, b3, out, ntiles);
}
